// PSMGNNModel_nolearn_75720273428993
// MI455X (gfx1250) — compile-verified, matrix operand order checked by reading, not run on hardware
//
#include <hip/hip_runtime.h>

// ============================================================================
// PSM-GNN belief propagation on MI455X (gfx1250, wave32, WMMA, async-to-LDS).
//
// Edge conv = sum over (kx,ky) taps of a 1D conv along Z as a Toeplitz GEMM:
//   D[m,r] = sum_s w[s-m] * In[s,r]   on V_WMMA_F32_16X16X32_BF16
//   M = 16 z-outputs (3 tiles), N = 16 y-rows, K = 48 padded to 64.
//
// This round: single-wave workgroups (no barriers at all on CDNA5 single-wave
// WGs), each wave owns 3 z-tiles x 4 batches = 12 f32 accumulators, so one
// 2 KB Toeplitz A tile feeds 24 WMMAs (A L2 traffic cut 12x vs round 1).
// Row staging is a 4-batch x 18-slot LDS ring fed by
// global_load_async_to_lds_b64 with counter-based pipelining
// (s_wait_asynccnt 4 -> 4 row copies always in flight under the WMMAs).
// Out-of-range rows async-copy from a zero page (no predicated ds_store path).
//
// ws: [mv 28.3MB f32][toep 35.7MB bf16][xbf 15MB][x1bf 15MB][zero page 256B]
// ============================================================================

#define NJ   17
#define NE   16
#define VD   48
#define KW   33
#define PADC 16
#define VOL  (VD * VD * VD)
#define RING 18                  // write@tap evicts a row last consumed 1 tap ago
#define ROWB 192                 // bytes per ring row: 16 zpad | 48 data | 32 zpad
#define RBYT (RING * ROWB)       // 3456 B per batch ring

typedef __attribute__((ext_vector_type(16))) __bf16       v16bf;
typedef __attribute__((ext_vector_type(8)))  float        v8f;
typedef __attribute__((ext_vector_type(4)))  float        v4f;
typedef __attribute__((ext_vector_type(4)))  unsigned int v4u;
typedef __attribute__((ext_vector_type(2)))  unsigned int v2u;

union BFrag {
    v16bf          v;
    unsigned short u[16];
    v4u            q[2];
};

__constant__ int c_parents[NE]  = {6, 6, 6, 2, 1, 3, 4, 7, 8, 8, 8, 9, 13, 12, 14, 15};
__constant__ int c_children[NE] = {2, 3, 7, 1, 0, 4, 5, 8, 9, 13, 14, 10, 12, 11, 15, 16};
__constant__ int c_isparent[NJ] = {0, 1, 1, 1, 1, 0, 1, 1, 1, 1, 0, 0, 1, 1, 1, 1, 0};

__device__ static inline unsigned short f2bf(float f) {
    unsigned int u = __float_as_uint(f);
    u += 0x7FFFu + ((u >> 16) & 1u);   // round-to-nearest-even
    return (unsigned short)(u >> 16);
}

// CDNA5 async global->LDS copy (ASYNCcnt) + counted waits, via inline asm
// (portable across ROCm 7.2 / amdgpu-toolchain clang builtin differences).
__device__ static inline void async_row_b64(unsigned lds_off, const void* gaddr) {
    asm volatile("global_load_async_to_lds_b64 %0, %1, off"
                 :: "v"(lds_off), "v"(gaddr)
                 : "memory");
}
__device__ static inline void wait_async4() {   // allow 4 copies in flight
    asm volatile("s_wait_asynccnt 0x4" ::: "memory");
}

// ----------------------------------------------------------------------------
// Kernel 1: fp32 -> bf16 volume conversion (4-wide).
// ----------------------------------------------------------------------------
__global__ __launch_bounds__(256) void cvt_bf16(const float* __restrict__ in,
                                                unsigned short* __restrict__ out,
                                                unsigned n4) {
    unsigned i = blockIdx.x * 256u + threadIdx.x;
    if (i >= n4) return;
    v4f f = *(const v4f*)(in + (size_t)i * 4);
    v2u p;
    p.x = (unsigned)f2bf(f.x) | ((unsigned)f2bf(f.y) << 16);
    p.y = (unsigned)f2bf(f.z) | ((unsigned)f2bf(f.w) << 16);
    *(v2u*)(out + (size_t)i * 4) = p;
}

// ----------------------------------------------------------------------------
// Kernel 2: Toeplitz A-fragments in the wave32 16-bit A-matrix layout
// (lanes 0-15: row m, K {0..7,16..23}; lanes 16-31: K {8..15,24..31}).
// Also zeroes the 256 B zero page (block 0, 2 halves per thread).
// ----------------------------------------------------------------------------
__global__ __launch_bounds__(64) void build_toeplitz(const float* __restrict__ w,
                                                     unsigned short* __restrict__ toep,
                                                     unsigned short* __restrict__ zpage) {
    int fid  = blockIdx.x;            // e*33*33 + kx*33 + ky
    int c    = threadIdx.x >> 5;      // K chunk: s in [32c, 32c+31]
    int lane = threadIdx.x & 31;
    int m    = lane & 15;
    int hi   = lane >> 4;

    if (blockIdx.x == 0) {            // zero page: 128 halves
        zpage[threadIdx.x]      = 0;
        zpage[threadIdx.x + 64] = 0;
    }

    const float* wrow = w + (size_t)fid * KW;   // w[e][kx][ky][0..32]

    BFrag f;
#pragma unroll
    for (int h = 0; h < 16; ++h) {
        int klocal = (h < 8 ? h : h + 8) + hi * 8;
        int d      = 32 * c + klocal - m;         // Toeplitz band index
        f.u[h]     = (d >= 0 && d < KW) ? f2bf(wrow[d]) : (unsigned short)0;
    }
    v4u* dst = (v4u*)(toep + ((size_t)fid * 2 + c) * 512) + lane * 2;
    dst[0] = f.q[0];
    dst[1] = f.q[1];
}

// ----------------------------------------------------------------------------
// Kernel 3: one grouped-conv pass. Single wave per WG: (edge, x-slice, y-tile);
// the wave carries 3 z-tiles x 4 batches (12 accumulators, 24 WMMAs per tap).
// mode 0: dst = mv[b,e] = conv+bias          (src joint = CHILDREN[e])
// mode 1: dst[b,CHILDREN[e]] *= (conv+bias)  (src joint = PARENTS[e])
// ----------------------------------------------------------------------------
__global__ __launch_bounds__(32) void edge_conv(const unsigned short* __restrict__ srcbf,
                                                const unsigned short* __restrict__ toep,
                                                const float* __restrict__ bias,
                                                const unsigned short* __restrict__ zpage,
                                                float* dst, int mode) {
    __shared__ __align__(16) unsigned short rings[4][RING][ROWB / 2];

    int e    = blockIdx.z;
    int x0   = blockIdx.x / 3;
    int y0   = (blockIdx.x % 3) * 16;
    int lane = threadIdx.x;
    int n    = lane & 15;
    int hi   = lane >> 4;

    int srcj = (mode == 0) ? c_children[e] : c_parents[e];
    const unsigned short* sv0 = srcbf + ((size_t)0 * NJ + srcj) * VOL;
    const unsigned short* sv1 = srcbf + ((size_t)1 * NJ + srcj) * VOL;
    const unsigned short* sv2 = srcbf + ((size_t)2 * NJ + srcj) * VOL;
    const unsigned short* sv3 = srcbf + ((size_t)3 * NJ + srcj) * VOL;

    // Zero the constant pad columns (halves [0,16) and [64,96) of every slot).
    unsigned short* flat = &rings[0][0][0];
    for (int i = lane; i < 4 * RING * 48; i += 32) {
        int bslot = i / 48, p = i % 48;
        flat[bslot * 96 + (p < 16 ? p : p + 48)] = 0;
    }

    const char* ringc = (const char*)flat;
    unsigned    lds0  = (unsigned)(unsigned long long)flat;
    bool        loader = (lane < 12);
    int         zt     = lane * 4;          // loader's z-chunk (halves), lane<12
    const unsigned short* zsrc = zpage + zt;

    v8f acc[12];
#pragma unroll
    for (int i = 0; i < 12; ++i) acc[i] = (v8f){0.f,0.f,0.f,0.f,0.f,0.f,0.f,0.f};

    for (int kx = 0; kx < KW; ++kx) {
        int xs = x0 + kx - PADC;
        if (xs < 0 || xs >= VD) continue;                 // uniform skip
        size_t so = (size_t)xs * (VD * VD) + zt;
        const unsigned short* s0 = sv0 + so;
        const unsigned short* s1 = sv1 + so;
        const unsigned short* s2 = sv2 + so;
        const unsigned short* s3 = sv3 + so;

        // Prime rows y0-16 .. y0 (17 rows x 4 batches; slot = (ys+36) % RING).
        if (loader) {
#pragma unroll 1
            for (int pr = 0; pr < 17; ++pr) {
                int      ys = y0 + pr - 16;
                unsigned a  = lds0 + (unsigned)(((ys + 36) % RING) * ROWB + 32 + lane * 8);
                int      ro = ys * VD;
                const unsigned short* p0 = (ys >= 0) ? s0 + ro : zsrc;
                const unsigned short* p1 = (ys >= 0) ? s1 + ro : zsrc;
                const unsigned short* p2 = (ys >= 0) ? s2 + ro : zsrc;
                const unsigned short* p3 = (ys >= 0) ? s3 + ro : zsrc;
                async_row_b64(a,            p0);
                async_row_b64(a + RBYT,     p1);
                async_row_b64(a + 2 * RBYT, p2);
                async_row_b64(a + 3 * RBYT, p3);
            }
        }
        wait_async4();   // everything but row y0 (needed first at tap 1) landed

        int bbase = ((y0 + n + 2) % RING) * ROWB;   // lane's B row: y0+n+ky-16
        int sbase = ((y0 + 1) % RING) * ROWB;       // store row:    y0+ky+1
        const char* tf = (const char*)(toep + (((size_t)e * KW + kx) * KW) * 1024);

#pragma unroll 1
        for (int ky = 0; ky < KW; ++ky) {
            // Stage row y0+ky+1 (first consumed at tap ky+2); evicts a row
            // last consumed at tap ky-1 -> safe without any barrier.
            if (loader) {
                int      ys = y0 + ky + 1;
                unsigned a  = lds0 + (unsigned)(sbase + 32 + lane * 8);
                int      ro = ys * VD;
                const unsigned short* p0 = (ys < VD) ? s0 + ro : zsrc;
                const unsigned short* p1 = (ys < VD) ? s1 + ro : zsrc;
                const unsigned short* p2 = (ys < VD) ? s2 + ro : zsrc;
                const unsigned short* p3 = (ys < VD) ? s3 + ro : zsrc;
                async_row_b64(a,            p0);
                async_row_b64(a + RBYT,     p1);
                async_row_b64(a + 2 * RBYT, p2);
                async_row_b64(a + 3 * RBYT, p3);
            }

            // One 2 KB Toeplitz tile feeds 24 WMMAs (4 batches x 3 z-tiles).
            BFrag A0, A1;
            const v4u* a0p = (const v4u*)tf + lane * 2;
            const v4u* a1p = (const v4u*)(tf + 512) + lane * 2;
            A0.q[0] = a0p[0]; A0.q[1] = a0p[1];
            A1.q[0] = a1p[0]; A1.q[1] = a1p[1];

#pragma unroll
            for (int c = 0; c < 2; ++c) {
#pragma unroll
                for (int bb = 0; bb < 4; ++bb) {
                    const char* rp = ringc + bb * RBYT + bbase + c * 64 + hi * 16;
#pragma unroll
                    for (int t = 0; t < 3; ++t) {
                        BFrag B;
                        B.q[0] = *(const v4u*)(rp + 32 * t);
                        B.q[1] = *(const v4u*)(rp + 32 * t + 32);
                        acc[bb * 3 + t] = __builtin_amdgcn_wmma_f32_16x16x32_bf16(
                            false, (c ? A1.v : A0.v), false, B.v,
                            (short)0, acc[bb * 3 + t], false, false);
                    }
                }
            }

            bbase += ROWB; if (bbase >= RBYT) bbase -= RBYT;
            sbase += ROWB; if (sbase >= RBYT) sbase -= RBYT;
            tf += 2048;

            wait_async4();   // keep exactly the 4 just-issued copies in flight
        }
    }

    // Epilogue. C/D layout: VGPR i -> (M = i or i+8 by lane half, N = lane&15).
    float bv   = bias[e];
    int   dstj = (mode == 0) ? 0 : c_children[e];
#pragma unroll
    for (int bb = 0; bb < 4; ++bb) {
#pragma unroll
        for (int t = 0; t < 3; ++t) {
#pragma unroll
            for (int i = 0; i < 8; ++i) {
                int   z    = 16 * t + ((lane < 16) ? i : i + 8);
                int   y    = y0 + n;
                float vout = acc[bb * 3 + t][i] + bv;
                if (mode == 0) {
                    dst[(((size_t)bb * NE + e) * VD + x0) * (VD * VD) +
                        (size_t)y * VD + z] = vout;
                } else {
                    size_t oidx = (((size_t)bb * NJ + dstj) * VD + x0) * (size_t)(VD * VD) +
                                  (size_t)y * VD + z;
                    dst[oidx] = dst[oidx] * vout;   // each element owned by one lane
                }
            }
        }
    }
}

// ----------------------------------------------------------------------------
// Kernel 4: pass-1 combine. x1 = parent? x*prod(messages) : (occ? 0.5x : x).
// Writes d_out (fp32; pass 2 RMWs the 16 child joints) and x1bf (pass-2 input).
// ----------------------------------------------------------------------------
__global__ __launch_bounds__(256) void combine_pass1(const float* __restrict__ x,
                                                     const int* __restrict__ occ,
                                                     const float* __restrict__ mv,
                                                     unsigned short* __restrict__ x1bf,
                                                     float* __restrict__ out) {
    size_t idx   = (size_t)blockIdx.x * blockDim.x + threadIdx.x;
    size_t total = (size_t)4 * NJ * VOL;
    if (idx >= total) return;

    int   p  = (int)(idx % VOL);
    int   j  = (int)((idx / VOL) % NJ);
    int   b  = (int)(idx / ((size_t)NJ * VOL));
    float xv = x[idx];
    float r;
    if (c_isparent[j]) {
        float prod = 1.0f;
#pragma unroll
        for (int e = 0; e < NE; ++e)
            if (c_parents[e] == j) prod *= mv[((size_t)b * NE + e) * VOL + p];
        r = xv * prod;
    } else {
        r = (occ[b * NJ + j] == 1) ? 0.5f * xv : xv;
    }
    out[idx]  = r;
    x1bf[idx] = f2bf(r);
}

// ----------------------------------------------------------------------------
extern "C" void kernel_launch(void* const* d_in, const int* in_sizes, int n_in,
                              void* d_out, int out_size, void* d_ws, size_t ws_size,
                              hipStream_t stream) {
    const float* x    = (const float*)d_in[0];
    const int*   occ  = (const int*)d_in[1];
    const float* w    = (const float*)d_in[2];
    const float* bias = (const float*)d_in[3];
    float*       out  = (float*)d_out;

    char*        ws         = (char*)d_ws;
    const size_t MV_BYTES   = (size_t)4 * NE * VOL * sizeof(float);           // 28.3 MB
    const size_t TOEP_BYTES = (size_t)NE * KW * KW * 2 * 512 * sizeof(short); // 35.7 MB
    const size_t XBF_BYTES  = (size_t)4 * NJ * VOL * sizeof(short);           // 15.0 MB
    float*          mv    = (float*)ws;
    unsigned short* toep  = (unsigned short*)(ws + MV_BYTES);
    unsigned short* xbf   = (unsigned short*)(ws + MV_BYTES + TOEP_BYTES);
    unsigned short* x1bf  = (unsigned short*)(ws + MV_BYTES + TOEP_BYTES + XBF_BYTES);
    unsigned short* zpage = (unsigned short*)(ws + MV_BYTES + TOEP_BYTES + 2 * XBF_BYTES);

    const size_t total = (size_t)4 * NJ * VOL;
    const unsigned n4  = (unsigned)(total / 4);

    // (1) Toeplitz A-fragments + zero page + bf16 copy of x
    build_toeplitz<<<NE * KW * KW, 64, 0, stream>>>(w, toep, zpage);
    cvt_bf16<<<(n4 + 255) / 256, 256, 0, stream>>>(x, xbf, n4);

    dim3 cgrid(VD * 3, 1, NE);   // (x-slice * y-tile, 1, edge); batches in-wave

    // (2) pass 1: leaf->root messages into mv
    edge_conv<<<cgrid, 32, 0, stream>>>(xbf, toep, bias, zpage, mv, 0);

    // (3) combine: occlusion + per-parent products -> d_out (f32) + x1bf
    combine_pass1<<<(unsigned)((total + 255) / 256), 256, 0, stream>>>(x, occ, mv, x1bf, out);

    // (4) pass 2: root->leaf, fused child multiply (RMW on d_out)
    edge_conv<<<cgrid, 32, 0, stream>>>(x1bf, toep, bias, zpage, out, 1);
}
